// MessagePassingLayer_14620068675791
// MI455X (gfx1250) — matrix operand order repaired and checked by reading, not compiled, not run on hardware
//
#include <hip/hip_runtime.h>
#include <hip/hip_bf16.h>

typedef __attribute__((ext_vector_type(16))) __bf16 bf16x16;
typedef __attribute__((ext_vector_type(8)))  __bf16 bf16x8;
typedef __attribute__((ext_vector_type(2)))  __bf16 bf16x2;
typedef __attribute__((ext_vector_type(8)))  float  f32x8;

// ---------- helpers ----------
__device__ __forceinline__ unsigned short f2bf(float f) {
    return __builtin_bit_cast(unsigned short, (__bf16)f);   // native RNE convert
}
__device__ __forceinline__ unsigned pack2bf(float x, float y) {
    bf16x2 t;
    t[0] = (__bf16)x;
    t[1] = (__bf16)y;
    return __builtin_bit_cast(unsigned, t);                 // v_cvt_pk_bf16_f32 path
}

__device__ __forceinline__ float leaky(float v) {
    return v >= 0.0f ? v : 0.01f * v;
}

// Load one A-operand 16-lane slice with compile-time-constant element offset.
// ISA A layout (16-bit A 16x32): k = e + 8*half + (e>=8 ? 8 : 0) => two 16B groups.
// base already includes row*stride + half*8; off = mt*16*stride + kc*32 (elements).
__device__ __forceinline__ bf16x16 load_a(const unsigned short* base, int off) {
    union { bf16x16 v; bf16x8 h[2]; } u;
    u.h[0] = *reinterpret_cast<const bf16x8*>(base + off);
    u.h[1] = *reinterpret_cast<const bf16x8*>(base + off + 16);
    return u.v;
}

// ---------- kernel: zero fill ----------
__global__ void fill_zero(float* p, size_t n) {
    size_t stride = (size_t)gridDim.x * blockDim.x;
    for (size_t i = (size_t)blockIdx.x * blockDim.x + threadIdx.x; i < n; i += stride)
        p[i] = 0.0f;
}

// ---------- kernel: weight prep (fp32 -> bf16, WMMA B-tile swizzle) ----------
// dst element index = ((nt*ktiles + kc)*32 + lane)*16 + e
// B layout (16-bit B KxN): k = e + 8*half + (e>=8 ? 8 : 0), matching the A layout.
__global__ void prep_weights(const float* __restrict__ W, unsigned short* __restrict__ dst,
                             int ktiles, int Ncols) {
    int idx  = blockIdx.x * 256 + threadIdx.x;
    int e    = idx & 15;
    int lane = (idx >> 4) & 31;
    int tl   = idx >> 9;
    int kc   = tl % ktiles;
    int nt   = tl / ktiles;
    int half = lane >> 4;
    int n    = nt * 16 + (lane & 15);
    int k    = kc * 32 + e + 8 * half + (e >= 8 ? 8 : 0);
    dst[idx] = f2bf(W[(size_t)k * Ncols + n]);
}

// ---------- kernel: per-graph row sums ----------
__global__ void reduce_rows(const float* __restrict__ src, float* __restrict__ dst,
                            int rowsPerGraph, int chunks) {
    const int ROWS = 2048;
    int g = blockIdx.x / chunks;
    int c = blockIdx.x % chunks;
    int start = g * rowsPerGraph + c * ROWS;
    int end   = g * rowsPerGraph + rowsPerGraph;
    if (start + ROWS < end) end = start + ROWS;
    int col = threadIdx.x & 127;
    int rl  = threadIdx.x >> 7;   // 0 or 1
    float acc = 0.0f;
    for (int row = start + rl; row < end; row += 2)
        acc += src[(size_t)row * 128 + col];
    atomicAdd(&dst[g * 128 + col], acc);
}

// ---------- kernel: per-edge MLPs via WMMA (64 edges / block, 4x weight reuse) ----------
#define XSTRIDE 520   // elements per sX row (512 + 8 pad -> 4-dword bank shift, 16B aligned)
#define HSTRIDE 264   // elements per sH row (256 + 8 pad)

__global__ void __launch_bounds__(256)
edge_kernel(const float* __restrict__ nodes, const float* __restrict__ edges,
            const float* __restrict__ glbs,
            const int* __restrict__ senders, const int* __restrict__ receivers,
            const float* __restrict__ bn0, const float* __restrict__ bn1,
            const float* __restrict__ be0, const float* __restrict__ be1,
            const unsigned short* __restrict__ wn0b, const unsigned short* __restrict__ wn1b,
            const unsigned short* __restrict__ we0b, const unsigned short* __restrict__ we1b,
            float* __restrict__ out_nodes, float* __restrict__ out_edges, int epg) {
    // 64 edges per block; concat row = 512 features (bf16).
    __shared__ unsigned short sX[64 * XSTRIDE];
    __shared__ unsigned short sH[2 * 64 * HSTRIDE];
    __shared__ int sSend[64];
    __shared__ int sRecv[64];

    const int tid  = threadIdx.x;
    const int base = blockIdx.x * 64;

    if (tid < 64) {
        sSend[tid] = senders[base + tid];
        sRecv[tid] = receivers[base + tid];
    }
    __syncthreads();

    // ---- Phase 1: build bf16 concat tile [64 x 512] in LDS ----
    // w = tid + i*256 => (w & 127) is loop-invariant: col/seg fixed per thread.
    {
        const int g4  = tid & 127;
        const int col = g4 * 4;
        const int seg = col >> 7;      // 0:sender 1:receiver 2:edge 3:global
        const int c   = col & 127;
        const int rB  = tid >> 7;      // 0 or 1; rows rB, rB+2, ..., rB+62
        #pragma unroll 4
        for (int i = 0; i < 32; ++i) {
            int row = rB + i * 2;
            const float* src;
            if (seg == 0)      src = nodes + (size_t)sSend[row] * 128 + c;
            else if (seg == 1) src = nodes + (size_t)sRecv[row] * 128 + c;
            else if (seg == 2) src = edges + (size_t)(base + row) * 128 + c;
            else               src = glbs  + (size_t)((base + row) / epg) * 128 + c;
            float4 v = *reinterpret_cast<const float4*>(src);
            uint2 pk;
            pk.x = pack2bf(v.x, v.y);
            pk.y = pack2bf(v.z, v.w);
            *reinterpret_cast<uint2*>(&sX[row * XSTRIDE + col]) = pk;
        }
    }
    __syncthreads();

    const int wave = tid >> 5;
    const int lane = tid & 31;
    const int half = lane >> 4;
    const int ln   = lane & 15;
    const int p    = wave >> 2;     // 0 = node-message MLP, 1 = edge-update MLP
    const int q    = wave & 3;

    const unsigned short* W0 = p ? we0b : wn0b;
    const unsigned short* W1 = p ? we1b : wn1b;
    const float*          b0 = p ? be0  : bn0;
    const float*          b1 = p ? be1  : bn1;

    // Per-wave A base pointers: lane-dependent part folded once; all load offsets
    // below are compile-time constants that fit the ds_load immediate field.
    const unsigned short* xB = &sX[ln * XSTRIDE + half * 8];
    const unsigned short* hB = &sH[(p * 64 + ln) * HSTRIDE + half * 8];

    // ---- Layer 1: [64x512] @ [512x256] -> hidden [64x256] (bf16 in LDS) ----
    #pragma unroll
    for (int t = 0; t < 4; ++t) {
        int nt = q * 4 + t;
        f32x8 acc0 = {}, acc1 = {}, acc2 = {}, acc3 = {};
        const bf16x16* wt = reinterpret_cast<const bf16x16*>(W0 + (size_t)(nt * 16) * 512);
        #pragma unroll
        for (int kc = 0; kc < 16; ++kc) {
            bf16x16 b = wt[kc * 32 + lane];
            bf16x16 a0 = load_a(xB, 0 * 16 * XSTRIDE + kc * 32);
            acc0 = __builtin_amdgcn_wmma_f32_16x16x32_bf16(false, a0, false, b, (short)0, acc0, false, false);
            bf16x16 a1 = load_a(xB, 1 * 16 * XSTRIDE + kc * 32);
            acc1 = __builtin_amdgcn_wmma_f32_16x16x32_bf16(false, a1, false, b, (short)0, acc1, false, false);
            bf16x16 a2 = load_a(xB, 2 * 16 * XSTRIDE + kc * 32);
            acc2 = __builtin_amdgcn_wmma_f32_16x16x32_bf16(false, a2, false, b, (short)0, acc2, false, false);
            bf16x16 a3 = load_a(xB, 3 * 16 * XSTRIDE + kc * 32);
            acc3 = __builtin_amdgcn_wmma_f32_16x16x32_bf16(false, a3, false, b, (short)0, acc3, false, false);
        }
        int   col = nt * 16 + ln;
        float bb  = b0[col];
        unsigned short* hDst = &sH[(p * 64) * HSTRIDE + col];
        #pragma unroll
        for (int r = 0; r < 8; ++r) {
            int ro = r + 8 * half;
            hDst[(0 * 16 + ro) * HSTRIDE] = f2bf(leaky(acc0[r] + bb));
            hDst[(1 * 16 + ro) * HSTRIDE] = f2bf(leaky(acc1[r] + bb));
            hDst[(2 * 16 + ro) * HSTRIDE] = f2bf(leaky(acc2[r] + bb));
            hDst[(3 * 16 + ro) * HSTRIDE] = f2bf(leaky(acc3[r] + bb));
        }
    }
    __syncthreads();

    // ---- Layer 2: [64x256] @ [256x128] -> out [64x128] ----
    #pragma unroll
    for (int t = 0; t < 2; ++t) {
        int nt = q + t * 4;
        f32x8 acc0 = {}, acc1 = {}, acc2 = {}, acc3 = {};
        const bf16x16* wt = reinterpret_cast<const bf16x16*>(W1 + (size_t)(nt * 8) * 512);
        #pragma unroll
        for (int kc = 0; kc < 8; ++kc) {
            bf16x16 b = wt[kc * 32 + lane];
            bf16x16 a0 = load_a(hB, 0 * 16 * HSTRIDE + kc * 32);
            acc0 = __builtin_amdgcn_wmma_f32_16x16x32_bf16(false, a0, false, b, (short)0, acc0, false, false);
            bf16x16 a1 = load_a(hB, 1 * 16 * HSTRIDE + kc * 32);
            acc1 = __builtin_amdgcn_wmma_f32_16x16x32_bf16(false, a1, false, b, (short)0, acc1, false, false);
            bf16x16 a2 = load_a(hB, 2 * 16 * HSTRIDE + kc * 32);
            acc2 = __builtin_amdgcn_wmma_f32_16x16x32_bf16(false, a2, false, b, (short)0, acc2, false, false);
            bf16x16 a3 = load_a(hB, 3 * 16 * HSTRIDE + kc * 32);
            acc3 = __builtin_amdgcn_wmma_f32_16x16x32_bf16(false, a3, false, b, (short)0, acc3, false, false);
        }
        int   col = nt * 16 + ln;
        float bb  = b1[col];
        if (p == 0) {
            #pragma unroll
            for (int r = 0; r < 8; ++r) {
                int ro = r + 8 * half;
                atomicAdd(&out_nodes[(size_t)sRecv[ 0 + ro] * 128 + col], leaky(acc0[r] + bb));
                atomicAdd(&out_nodes[(size_t)sRecv[16 + ro] * 128 + col], leaky(acc1[r] + bb));
                atomicAdd(&out_nodes[(size_t)sRecv[32 + ro] * 128 + col], leaky(acc2[r] + bb));
                atomicAdd(&out_nodes[(size_t)sRecv[48 + ro] * 128 + col], leaky(acc3[r] + bb));
            }
        } else {
            #pragma unroll
            for (int r = 0; r < 8; ++r) {
                int ro = r + 8 * half;
                out_edges[(size_t)(base +  0 + ro) * 128 + col] = leaky(acc0[r] + bb);
                out_edges[(size_t)(base + 16 + ro) * 128 + col] = leaky(acc1[r] + bb);
                out_edges[(size_t)(base + 32 + ro) * 128 + col] = leaky(acc2[r] + bb);
                out_edges[(size_t)(base + 48 + ro) * 128 + col] = leaky(acc3[r] + bb);
            }
        }
    }
}

// ---------- kernel: global update (tiny) ----------
__global__ void global_final(const float* __restrict__ glbs,
                             const float* __restrict__ nsum, const float* __restrict__ esum,
                             const float* __restrict__ Wgn, const float* __restrict__ bgn,
                             const float* __restrict__ Wge, const float* __restrict__ bge,
                             const float* __restrict__ Wg,  const float* __restrict__ bg,
                             const float* __restrict__ Wf,  const float* __restrict__ bf,
                             float* __restrict__ outg, int G) {
    __shared__ float tmp[8][384];
    int col = threadIdx.x;      // 128 threads
    for (int g = 0; g < G; ++g) {
        float ag = bg[col], an = bgn[col], ae = bge[col];
        for (int k = 0; k < 128; ++k) {
            ag += glbs[g * 128 + k] * Wg[k * 128 + col];
            an += nsum[g * 128 + k] * Wgn[k * 128 + col];
            ae += esum[g * 128 + k] * Wge[k * 128 + col];
        }
        tmp[g][col]       = leaky(ag);
        tmp[g][128 + col] = leaky(an);
        tmp[g][256 + col] = leaky(ae);
    }
    __syncthreads();
    for (int g = 0; g < G; ++g) {
        float o = bf[col];
        for (int k = 0; k < 384; ++k)
            o += tmp[g][k] * Wf[k * 128 + col];
        outg[g * 128 + col] = leaky(o);
    }
}

// ---------- host launcher ----------
extern "C" void kernel_launch(void* const* d_in, const int* in_sizes, int n_in,
                              void* d_out, int out_size, void* d_ws, size_t ws_size,
                              hipStream_t stream) {
    const float* nodes     = (const float*)d_in[0];
    const float* edges     = (const float*)d_in[1];
    const float* glbs      = (const float*)d_in[2];
    const int*   senders   = (const int*)d_in[3];
    const int*   receivers = (const int*)d_in[4];
    const float* Wn0 = (const float*)d_in[7];
    const float* bn0 = (const float*)d_in[8];
    const float* Wn1 = (const float*)d_in[9];
    const float* bn1 = (const float*)d_in[10];
    const float* We0 = (const float*)d_in[11];
    const float* be0 = (const float*)d_in[12];
    const float* We1 = (const float*)d_in[13];
    const float* be1 = (const float*)d_in[14];
    const float* Wgn = (const float*)d_in[15];
    const float* bgn = (const float*)d_in[16];
    const float* Wge = (const float*)d_in[17];
    const float* bge = (const float*)d_in[18];
    const float* Wg  = (const float*)d_in[19];
    const float* bg  = (const float*)d_in[20];
    const float* Wf  = (const float*)d_in[21];
    const float* bf  = (const float*)d_in[22];

    const int N = in_sizes[0] / 128;
    const int E = in_sizes[1] / 128;
    const int G = in_sizes[2] / 128;

    // workspace layout
    unsigned short* wn0b = (unsigned short*)d_ws;        // 512*256 bf16
    unsigned short* we0b = wn0b + 512 * 256;
    unsigned short* wn1b = we0b + 512 * 256;             // 256*128 bf16
    unsigned short* we1b = wn1b + 256 * 128;
    float* nsum = (float*)(we1b + 256 * 128);            // G*128
    float* esum = nsum + G * 128;

    float* out_nodes = (float*)d_out;
    float* out_edges = out_nodes + (size_t)N * 128;
    float* out_glb   = out_edges + (size_t)E * 128;

    // 1) zero scatter targets + accumulators
    fill_zero<<<1024, 256, 0, stream>>>(out_nodes, (size_t)N * 128);
    fill_zero<<<8, 256, 0, stream>>>(nsum, (size_t)2 * G * 128);

    // 2) weight prep: fp32 -> bf16 WMMA-B tiles
    prep_weights<<<512, 256, 0, stream>>>(Wn0, wn0b, 16, 256);
    prep_weights<<<512, 256, 0, stream>>>(We0, we0b, 16, 256);
    prep_weights<<<128, 256, 0, stream>>>(Wn1, wn1b, 8, 128);
    prep_weights<<<128, 256, 0, stream>>>(We1, we1b, 8, 128);

    // 3) per-graph feature sums
    int npg = N / G, epg = E / G;
    int nchunks = (npg + 2047) / 2048;
    int echunks = (epg + 2047) / 2048;
    reduce_rows<<<G * nchunks, 256, 0, stream>>>(nodes, nsum, npg, nchunks);
    reduce_rows<<<G * echunks, 256, 0, stream>>>(edges, esum, epg, echunks);

    // 4) main edge MLPs + scatter (64 edges per block)
    edge_kernel<<<E / 64, 256, 0, stream>>>(nodes, edges, glbs, senders, receivers,
                                            bn0, bn1, be0, be1,
                                            wn0b, wn1b, we0b, we1b,
                                            out_nodes, out_edges, epg);

    // 5) global update
    global_final<<<1, 128, 0, stream>>>(glbs, nsum, esum,
                                        Wgn, bgn, Wge, bge, Wg, bg, Wf, bf,
                                        out_glb, G);
}
